// Transformer_77807627534769
// MI455X (gfx1250) — matrix operand order repaired and checked by reading, not compiled, not run on hardware
//
#include <hip/hip_runtime.h>
#include <math.h>

typedef __attribute__((ext_vector_type(16))) __bf16 bf16x16;
typedef __attribute__((ext_vector_type(8)))  float  f32x8;

#define SEQ   2049
#define DIMC  512
#define INNERC 1024
#define QKVC  3072
#define NTOK  2048

__device__ __forceinline__ __bf16 to_bf16(float f) { return (__bf16)f; }

union PK4 { __bf16 h[4]; uint2 u; };

// ---------------------------------------------------------------------------
// Generic GEMM: C[M x N] = epi(A[M x K] @ W[K x N] + bias)
// bf16 WMMA 16x16x32, f32 accumulate. Block: 256 threads = 8 waves,
// tile 64(M) x 64(N); waves arranged 4(m) x 2(n); each wave keeps TWO
// accumulators (B tiles 2*wn, 2*wn+1) reusing its A fragment.
// LDS tiles stored in WMMA *fragment order* so fragments load as one
// contiguous 32-byte ds_load vector per lane:
//   A-frag (16x32 bf16): lane = m + kHalf*16 ; elem e -> k = e + kHalf*16
//   B-frag (32x16 bf16): lane = n + kHalf*16 ; elem e -> k = e + kHalf*16
// A staging: a 4-aligned k-chunk stays inside one octet -> same lane row,
// consecutive e, so 4 bf16 pack into one 8-byte LDS store.
// epi: 0=none 1=relu 2=gelu(exact). remap: out_row = b*(nb+1)+1+r%nb (cls gap)
// ---------------------------------------------------------------------------
__global__ void gemm_wmma_kernel(const float* __restrict__ A,
                                 const float* __restrict__ W,
                                 const float* __restrict__ bias,
                                 float* __restrict__ C,
                                 int M, int K, int N,
                                 int epi, int remap, int nb) {
    __shared__ alignas(32) __bf16 AsmF[4][32][16];   // [mTile][lane][elem]
    __shared__ alignas(32) __bf16 BsmF[4][32][16];   // [nTile][lane][elem]

    const int tid  = threadIdx.x;
    const int lane = tid & 31;
    const int wave = tid >> 5;
    const int wm   = wave & 3;          // 0..3 : 16-row tile within 64
    const int wn   = wave >> 2;         // 0..1 : pair of 16-col tiles within 64
    const int mbase = blockIdx.y * 64;
    const int nbase = blockIdx.x * 64;

    f32x8 acc0 = {0.f,0.f,0.f,0.f,0.f,0.f,0.f,0.f};
    f32x8 acc1 = {0.f,0.f,0.f,0.f,0.f,0.f,0.f,0.f};

    for (int k0 = 0; k0 < K; k0 += 32) {
        __syncthreads();
        // stage A tile 64x32 (512 float4 / 256 thr = 2 iters)
#pragma unroll
        for (int i = tid; i < 64 * 8; i += 256) {
            int r = i >> 3, c0 = (i & 7) * 4;
            int grow = mbase + r;
            float4 av = make_float4(0.f, 0.f, 0.f, 0.f);
            if (grow < M) av = *(const float4*)&A[(size_t)grow * K + k0 + c0];
            int hf = (c0 >> 4) & 1;
            int e0 = c0 & 15;
            PK4 pk;
            pk.h[0] = to_bf16(av.x); pk.h[1] = to_bf16(av.y);
            pk.h[2] = to_bf16(av.z); pk.h[3] = to_bf16(av.w);
            *(uint2*)&AsmF[r >> 4][(r & 15) + hf * 16][e0] = pk.u;
        }
        // stage B tile 32x64 (512 float4 / 256 thr = 2 iters)
#pragma unroll
        for (int i = tid; i < 32 * 16; i += 256) {
            int r = i >> 4, c0 = (i & 15) * 4;       // r = k, c0 = n start
            float4 wv = *(const float4*)&W[(size_t)(k0 + r) * N + nbase + c0];
            int hf = r >> 4, e = r & 15;
            int ntile = c0 >> 4;                     // constant across the 4
            float wa[4] = {wv.x, wv.y, wv.z, wv.w};
#pragma unroll
            for (int j = 0; j < 4; j++)
                BsmF[ntile][((c0 + j) & 15) + hf * 16][e] = to_bf16(wa[j]);
        }
        // prefetch next K-step tiles while we compute
        if (k0 + 32 < K) {
            int r = tid >> 5, c = tid & 31;
            int grow = mbase + r * 8;                // spread over the 64 rows
            if (grow < M)
                __builtin_prefetch(&A[(size_t)grow * K + k0 + 32 + c], 0, 3);
            __builtin_prefetch(&W[(size_t)(k0 + 32 + r) * N + nbase + c * 2], 0, 3);
        }
        __syncthreads();
        bf16x16 afr  = *(const bf16x16*)&AsmF[wm][lane][0];
        bf16x16 bfr0 = *(const bf16x16*)&BsmF[2 * wn + 0][lane][0];
        bf16x16 bfr1 = *(const bf16x16*)&BsmF[2 * wn + 1][lane][0];
        acc0 = __builtin_amdgcn_wmma_f32_16x16x32_bf16(
                   false, afr, false, bfr0, (short)0, acc0, false, false);
        acc1 = __builtin_amdgcn_wmma_f32_16x16x32_bf16(
                   false, afr, false, bfr1, (short)0, acc1, false, false);
    }

    // epilogue: C-layout  (m = r + 8*half, n = lane&15)
    const int half = lane >> 4, nl = lane & 15;
#pragma unroll
    for (int s = 0; s < 2; s++) {
        const f32x8 acc = s ? acc1 : acc0;
        const int gcol = nbase + (2 * wn + s) * 16 + nl;
        const float bv = bias ? bias[gcol] : 0.f;
#pragma unroll
        for (int r = 0; r < 8; r++) {
            int grow = mbase + wm * 16 + r + 8 * half;
            if (grow >= M) continue;
            float v = acc[r] + bv;
            if (epi == 1)      v = fmaxf(v, 0.f);
            else if (epi == 2) v = 0.5f * v * (1.f + erff(v * 0.70710678118654752440f));
            int orow = remap ? ((grow / nb) * (nb + 1) + 1 + (grow % nb)) : grow;
            C[(size_t)orow * N + gcol] = v;
        }
    }
}

// ---------------------------------------------------------------------------
// Flash attention: 4 waves per block, each wave owns one 16-row query tile;
// the 32-key x 128-d K and V blocks are staged cooperatively and reused by
// all 4 waves. Per key block & wave: 8 WMMA for S = Q K^T, online softmax,
// 8 WMMA for O += P V.
// QKV layout: [b, n, 3*INNER] with q|k|v sections, col = h*128 + d.
// ---------------------------------------------------------------------------
__global__ void attn_kernel(const float* __restrict__ qkv,
                            const float* __restrict__ mask,
                            float* __restrict__ O) {
    const int tid  = threadIdx.x;            // 0..127
    const int lane = tid & 31;
    const int wave = tid >> 5;               // 0..3
    const int half = lane >> 4, nl = lane & 15;
    const int qt = blockIdx.x * 4 + wave;    // 16-row query tile index
    const int h  = blockIdx.y, b = blockIdx.z;

    __shared__ alignas(32) __bf16 Ksm[4][2][32][16]; // [kChunk][keyTile][lane][e]
    __shared__ alignas(32) __bf16 Vsm[8][32][16];    // [dTile][lane][e]
    __shared__ alignas(32) __bf16 Psm[4][32][16];    // per-wave A-frag of P

    const float scale = 0.08838834764831845f;        // 128^-0.5

    // ---- Q fragments (scaled), A-frag layout: m = nl, k-set depends on half
    bf16x16 qf[4];
    {
        int qm = qt * 16 + nl;
        int qrow = (qm < SEQ) ? qm : (SEQ - 1);
        const float* qbase = qkv + ((size_t)(b * SEQ + qrow)) * QKVC + h * 128;
#pragma unroll
        for (int c = 0; c < 4; c++)
#pragma unroll
            for (int e = 0; e < 16; e++) {
                int k = c * 32 + e + half * 16;
                qf[c][e] = to_bf16(qbase[k] * scale);
            }
    }

    f32x8 oacc[8];
#pragma unroll
    for (int dt = 0; dt < 8; dt++) oacc[dt] = (f32x8){0.f,0.f,0.f,0.f,0.f,0.f,0.f,0.f};
    float m_st[8], l_st[8];
#pragma unroll
    for (int r = 0; r < 8; r++) { m_st[r] = -1e30f; l_st[r] = 0.f; }

    const int nkb = (SEQ + 31) / 32;   // 65
    for (int jb = 0; jb < nkb; jb++) {
        const int kbase = jb * 32;
        __syncthreads();
        // ---- stage K and V blocks (32 keys x 128 dh) cooperatively (128 thr)
#pragma unroll
        for (int t = 0; t < 8; t++) {
            int fi = t * 128 + tid;           // float4 index, 0..1023
            int kr = fi >> 5;                 // key row 0..31
            int d0 = (fi & 31) * 4;           // dh start (4-aligned)
            int key = kbase + kr;
            float4 kv = make_float4(0.f,0.f,0.f,0.f);
            float4 vv = make_float4(0.f,0.f,0.f,0.f);
            if (key < SEQ) {
                const float* rp = qkv + ((size_t)(b * SEQ + key)) * QKVC + h * 128;
                kv = *(const float4*)(rp + INNERC + d0);       // K section
                vv = *(const float4*)(rp + 2 * INNERC + d0);   // V section
            }
            int tt = kr >> 4, km = kr & 15;
            { // K as B-frag of QK^T: same octet for the 4-chunk -> packed store
                int ch = d0 >> 5, hf = (d0 >> 4) & 1, e0 = d0 & 15;
                PK4 pk;
                pk.h[0] = to_bf16(kv.x); pk.h[1] = to_bf16(kv.y);
                pk.h[2] = to_bf16(kv.z); pk.h[3] = to_bf16(kv.w);
                *(uint2*)&Ksm[ch][tt][km + hf * 16][e0] = pk.u;
            }
            { // V as B-frag of PV: scatter across lanes (e = key, lane = d)
                float va[4] = {vv.x, vv.y, vv.z, vv.w};
                int hf = kr >> 4, e = kr & 15;
                int dt = d0 >> 4;             // constant across the 4
#pragma unroll
                for (int j = 0; j < 4; j++)
                    Vsm[dt][((d0 + j) & 15) + hf * 16][e] = to_bf16(va[j]);
            }
        }
        __syncthreads();

        // ---- S = Q K^T  (two 16x16 tiles over keys, K-dim 128 in 4 chunks)
        f32x8 S[2];
#pragma unroll
        for (int t = 0; t < 2; t++) {
            f32x8 s = {0.f,0.f,0.f,0.f,0.f,0.f,0.f,0.f};
#pragma unroll
            for (int c = 0; c < 4; c++) {
                bf16x16 bk = *(const bf16x16*)&Ksm[c][t][lane][0];
                s = __builtin_amdgcn_wmma_f32_16x16x32_bf16(
                        false, qf[c], false, bk, (short)0, s, false, false);
            }
            S[t] = s;
        }

        // ---- mask + online softmax (row = r + 8*half, col = lane&15)
        float keepv[2];
#pragma unroll
        for (int t = 0; t < 2; t++) {
            int key = kbase + t * 16 + nl;
            float mk = 0.f;
            if (key < SEQ) mk = (key == 0) ? 1.f : mask[(size_t)b * NTOK + key - 1];
            keepv[t] = mk;
        }
#pragma unroll
        for (int r = 0; r < 8; r++) {
            float s0 = (keepv[0] != 0.f) ? S[0][r] : -1e30f;
            float s1 = (keepv[1] != 0.f) ? S[1][r] : -1e30f;
            float mx = fmaxf(s0, s1);
            for (int off = 8; off >= 1; off >>= 1)
                mx = fmaxf(mx, __shfl_xor(mx, off, 32));
            float mnew  = fmaxf(m_st[r], mx);
            float alpha = expf(m_st[r] - mnew);
            float p0 = expf(s0 - mnew), p1 = expf(s1 - mnew);
            float rs = p0 + p1;
            for (int off = 8; off >= 1; off >>= 1)
                rs += __shfl_xor(rs, off, 32);
            l_st[r] = l_st[r] * alpha + rs;
            m_st[r] = mnew;
#pragma unroll
            for (int dt = 0; dt < 8; dt++) oacc[dt][r] = oacc[dt][r] * alpha;
            // write P into this wave's A-fragment: m = r+8*half, k = t*16 + nl
            int m = r + 8 * half;
#pragma unroll
            for (int t = 0; t < 2; t++) {
                int k = t * 16 + nl;
                int hfa = k >> 4;
                int e = k & 15;
                Psm[wave][m + hfa * 16][e] = to_bf16(t ? p1 : p0);
            }
        }
        __syncthreads();

        // ---- O += P V
        bf16x16 pa = *(const bf16x16*)&Psm[wave][lane][0];
#pragma unroll
        for (int dt = 0; dt < 8; dt++) {
            bf16x16 bv = *(const bf16x16*)&Vsm[dt][lane][0];
            oacc[dt] = __builtin_amdgcn_wmma_f32_16x16x32_bf16(
                           false, pa, false, bv, (short)0, oacc[dt], false, false);
        }
    }

    // ---- normalize + store (output layout [b, n, h*128 + d])
#pragma unroll
    for (int r = 0; r < 8; r++) {
        int row = qt * 16 + r + 8 * half;
        if (row >= SEQ) continue;
        float inv = 1.f / l_st[r];
#pragma unroll
        for (int dt = 0; dt < 8; dt++)
            O[((size_t)(b * SEQ + row)) * INNERC + h * 128 + dt * 16 + nl] =
                oacc[dt][r] * inv;
    }
}

// ---------------------------------------------------------------------------
// LayerNorm over last dim (512). One block (128 threads) per row.
// ---------------------------------------------------------------------------
__global__ void ln_kernel(const float* __restrict__ in,
                          const float* __restrict__ g,
                          const float* __restrict__ bta,
                          float* __restrict__ out) {
    const int row = blockIdx.x;
    const int tid = threadIdx.x;   // 0..127
    __shared__ float red[128];
    const float* x = in + (size_t)row * DIMC;
    float v4[4];
    float s = 0.f;
#pragma unroll
    for (int i = 0; i < 4; i++) { v4[i] = x[tid + i * 128]; s += v4[i]; }
    red[tid] = s; __syncthreads();
    for (int o = 64; o > 0; o >>= 1) { if (tid < o) red[tid] += red[tid + o]; __syncthreads(); }
    float mean = red[0] / (float)DIMC; __syncthreads();
    float vs = 0.f;
#pragma unroll
    for (int i = 0; i < 4; i++) { float d = v4[i] - mean; vs += d * d; }
    red[tid] = vs; __syncthreads();
    for (int o = 64; o > 0; o >>= 1) { if (tid < o) red[tid] += red[tid + o]; __syncthreads(); }
    float inv = rsqrtf(red[0] / (float)DIMC + 1e-5f);
#pragma unroll
    for (int i = 0; i < 4; i++) {
        int c = tid + i * 128;
        out[(size_t)row * DIMC + c] = (v4[i] - mean) * inv * g[c] + bta[c];
    }
}

// cls token -> row 0 of each batch
__global__ void cls_kernel(const float* __restrict__ cls, float* __restrict__ H) {
    H[(size_t)blockIdx.x * SEQ * DIMC + threadIdx.x] = cls[threadIdx.x];
}

// pooled = H[b,0]; double LN; logits = pooled @ head_w + head_b
__global__ void head_kernel(const float* __restrict__ H,
                            const float* __restrict__ ng, const float* __restrict__ nb_,
                            const float* __restrict__ hg, const float* __restrict__ hb,
                            const float* __restrict__ hw, const float* __restrict__ hbias,
                            float* __restrict__ out) {
    const int b = blockIdx.x;
    const int tid = threadIdx.x;   // 0..511
    __shared__ float red[512];
    float x = H[(size_t)b * SEQ * DIMC + tid];
#pragma unroll
    for (int pass = 0; pass < 2; pass++) {
        red[tid] = x; __syncthreads();
        for (int o = 256; o > 0; o >>= 1) { if (tid < o) red[tid] += red[tid + o]; __syncthreads(); }
        float mean = red[0] / (float)DIMC; __syncthreads();
        float d = x - mean;
        red[tid] = d * d; __syncthreads();
        for (int o = 256; o > 0; o >>= 1) { if (tid < o) red[tid] += red[tid + o]; __syncthreads(); }
        float inv = rsqrtf(red[0] / (float)DIMC + 1e-5f); __syncthreads();
        const float* gg = pass ? hg : ng;
        const float* bb = pass ? hb : nb_;
        x = d * inv * gg[tid] + bb[tid];
    }
    float s01[2];
#pragma unroll
    for (int c = 0; c < 2; c++) {
        red[tid] = x * hw[tid * 2 + c]; __syncthreads();
        for (int o = 256; o > 0; o >>= 1) { if (tid < o) red[tid] += red[tid + o]; __syncthreads(); }
        s01[c] = red[0]; __syncthreads();
    }
    if (tid == 0) { out[b * 2 + 0] = s01[0] + hbias[0]; out[b * 2 + 1] = s01[1] + hbias[1]; }
}

// ---------------------------------------------------------------------------
extern "C" void kernel_launch(void* const* d_in, const int* in_sizes, int n_in,
                              void* d_out, int out_size, void* d_ws, size_t ws_size,
                              hipStream_t stream) {
    const float* x         = (const float*)d_in[0];
    const float* mask      = (const float*)d_in[1];
    const float* proj_w    = (const float*)d_in[2];
    const float* proj_b    = (const float*)d_in[3];
    const float* cls_token = (const float*)d_in[4];
    const float* ln1_g     = (const float*)d_in[5];
    const float* ln1_b     = (const float*)d_in[6];
    const float* qkv_w     = (const float*)d_in[7];
    const float* out_w     = (const float*)d_in[8];
    const float* out_b     = (const float*)d_in[9];
    const float* ln2_g     = (const float*)d_in[10];
    const float* ln2_b     = (const float*)d_in[11];
    const float* ff_w1     = (const float*)d_in[12];
    const float* ff_b1     = (const float*)d_in[13];
    const float* ff_w2     = (const float*)d_in[14];
    const float* ff_b2     = (const float*)d_in[15];
    const float* norm_g    = (const float*)d_in[16];
    const float* norm_b    = (const float*)d_in[17];
    const float* head_ln_g = (const float*)d_in[18];
    const float* head_ln_b = (const float*)d_in[19];
    const float* head_w    = (const float*)d_in[20];
    const float* head_b    = (const float*)d_in[21];
    float* out = (float*)d_out;

    // workspace carve-out (256B aligned)
    char* ws = (char*)d_ws;
    size_t off = 0;
    auto alloc = [&](size_t nElem) {
        float* p = (float*)(ws + off);
        off += ((nElem * sizeof(float) + 255) / 256) * 256;
        return p;
    };
    float* H   = alloc((size_t)2 * SEQ * DIMC);    //  8.4 MB
    float* HN  = alloc((size_t)2 * SEQ * DIMC);    //  8.4 MB
    float* QKV = alloc((size_t)2 * SEQ * QKVC);    // 50.3 MB
    float* Ob  = alloc((size_t)2 * SEQ * INNERC);  // 16.8 MB
    float* F   = alloc((size_t)2 * SEQ * INNERC);  // 16.8 MB
    (void)ws_size; (void)in_sizes; (void)n_in; (void)out_size;

    const dim3 blk(256);
    const int rows  = 2 * SEQ;                 // 4098
    const int mtR   = (rows + 63) / 64;        // 65

    // input projection (+ReLU) into H rows 1..2048 per batch, cls into row 0
    cls_kernel<<<2, DIMC, 0, stream>>>(cls_token, H);
    gemm_wmma_kernel<<<dim3(DIMC / 64, 4096 / 64), blk, 0, stream>>>(
        x, proj_w, proj_b, H, 4096, 1024, DIMC, /*relu*/1, /*remap*/1, NTOK);

    for (int l = 0; l < 4; l++) {
        ln_kernel<<<rows, 128, 0, stream>>>(H, ln1_g + l * DIMC, ln1_b + l * DIMC, HN);
        gemm_wmma_kernel<<<dim3(QKVC / 64, mtR), blk, 0, stream>>>(
            HN, qkv_w + (size_t)l * DIMC * QKVC, nullptr, QKV,
            rows, DIMC, QKVC, 0, 0, 1);
        attn_kernel<<<dim3((SEQ + 63) / 64, 8, 2), 128, 0, stream>>>(QKV, mask, Ob);
        gemm_wmma_kernel<<<dim3(DIMC / 64, mtR), blk, 0, stream>>>(
            Ob, out_w + (size_t)l * INNERC * DIMC, out_b + l * DIMC, H,
            rows, INNERC, DIMC, 0, 0, 1);
        ln_kernel<<<rows, 128, 0, stream>>>(H, ln2_g + l * DIMC, ln2_b + l * DIMC, HN);
        gemm_wmma_kernel<<<dim3(INNERC / 64, mtR), blk, 0, stream>>>(
            HN, ff_w1 + (size_t)l * DIMC * INNERC, ff_b1 + l * INNERC, F,
            rows, DIMC, INNERC, /*gelu*/2, 0, 1);
        gemm_wmma_kernel<<<dim3(DIMC / 64, mtR), blk, 0, stream>>>(
            F, ff_w2 + (size_t)l * INNERC * DIMC, ff_b2 + l * DIMC, H,
            rows, INNERC, DIMC, 0, 0, 1);
    }

    head_kernel<<<2, DIMC, 0, stream>>>(H, norm_g, norm_b, head_ln_g, head_ln_b,
                                        head_w, head_b, out);
}
